// Net_64544768525120
// MI455X (gfx1250) — matrix operand order repaired and checked by reading, not compiled, not run on hardware
//
#include <hip/hip_runtime.h>
#include <hip/hip_bf16.h>

#define N_NODES   100000
#define N_FEAT    512
#define HIDDEN    16
#define N_CLASSES 7
#define N_EDGES   3200000

typedef __attribute__((ext_vector_type(16))) _Float16 v16h;
typedef __attribute__((ext_vector_type(8)))  float    v8f;

// ---------------------------------------------------------------- degree ----
__global__ __launch_bounds__(256) void k_init_deg(float* __restrict__ deg) {
    int i = blockIdx.x * 256 + threadIdx.x;
    if (i < N_NODES) deg[i] = 1.0f;              // self-loop contribution
}

__global__ __launch_bounds__(256) void k_deg_accum(const int* __restrict__ dst,
                                                   float* __restrict__ deg) {
    int e = blockIdx.x * 256 + threadIdx.x;
    if (e < N_EDGES) atomicAdd(&deg[dst[e]], 1.0f);
}

__global__ __launch_bounds__(256) void k_deg_rsqrt(float* __restrict__ deg) {
    int i = blockIdx.x * 256 + threadIdx.x;
    if (i < N_NODES) deg[i] = rsqrtf(deg[i]);    // deg >= 1, so always valid
}

// ------------------------------------------------------- layer-1 WMMA GEMM --
// h1[100000x16] = x[100000x512] @ W1[512x16], one 16x16 tile per wave,
// v_wmma_f32_16x16x32_f16, K=512 -> 16 WMMA per tile. W1 staged in LDS (f16).
__global__ __launch_bounds__(256) void k_gemm1_wmma(const float* __restrict__ x,
                                                    const float* __restrict__ W1,
                                                    float* __restrict__ h1) {
    __shared__ _Float16 sW[N_FEAT * HIDDEN];     // 16 KB
    for (int i = threadIdx.x; i < N_FEAT * HIDDEN; i += 256)
        sW[i] = (_Float16)W1[i];
    __syncthreads();

    const int wave = threadIdx.x >> 5;
    const int lane = threadIdx.x & 31;
    const int tile = blockIdx.x * 8 + wave;      // 16-row tile index
    if (tile >= N_NODES / 16) return;            // wave-uniform: EXEC stays full

    const int mrow = (lane & 15);                // A-matrix M = lane (mod 16)
    const int half = lane >> 4;                  // lane group selects K half
    const int ncol = lane & 15;                  // B/C/D column
    const float* xr = x + (size_t)(tile * 16 + mrow) * N_FEAT;

    v8f c = {};
    for (int k0 = 0; k0 < N_FEAT; k0 += 32) {
        // A fragment: 16x32 f16 per ISA 7.12.2 (pairs of K per VGPR)
        v16h a;
        #pragma unroll
        for (int v = 0; v < 8; ++v) {
            const int kb = k0 + ((v >> 2) << 4) + (half << 3) + ((v & 3) << 1);
            const float2 f2 = *(const float2*)(xr + kb);
            a[2 * v]     = (_Float16)f2.x;
            a[2 * v + 1] = (_Float16)f2.y;
        }
        // B fragment: 32x16 f16, rows striped across lanes, same K order as A
        v16h b;
        #pragma unroll
        for (int v = 0; v < 8; ++v) {
            const int k = k0 + ((v >> 2) << 4) + (half << 3) + ((v & 3) << 1);
            b[2 * v]     = sW[k * HIDDEN + ncol];
            b[2 * v + 1] = sW[(k + 1) * HIDDEN + ncol];
        }
        c = __builtin_amdgcn_wmma_f32_16x16x32_f16(
                false, a, false, b, (short)0, c, false, false);
    }

    // C/D layout: VGPR r -> M = r + 8*(lane>>4), N = lane&15
    #pragma unroll
    for (int r = 0; r < 8; ++r) {
        const int m = tile * 16 + r + (half << 3);
        h1[(size_t)m * HIDDEN + ncol] = c[r];
    }
}

// ------------------------------------------------- layer-1 self-loop init ---
__global__ __launch_bounds__(256) void k_init_agg1(const float* __restrict__ h1,
                                                   const float* __restrict__ dinv,
                                                   float* __restrict__ agg1) {
    int idx = blockIdx.x * 256 + threadIdx.x;
    if (idx < N_NODES * HIDDEN) {
        int n = idx >> 4;
        float d = dinv[n];
        agg1[idx] = h1[idx] * d * d;             // self-loop: norm = dinv^2
    }
}

// --------------------------------------------------- layer-1 edge scatter ---
__global__ __launch_bounds__(256) void k_scatter1(const int* __restrict__ src,
                                                  const int* __restrict__ dst,
                                                  const float* __restrict__ dinv,
                                                  const float* __restrict__ h1,
                                                  float* __restrict__ agg1) {
    int e = blockIdx.x * 256 + threadIdx.x;
    if (e >= N_EDGES) return;
    const int s = src[e], d = dst[e];
    const float w = dinv[s] * dinv[d];
    const float4* hp = (const float4*)(h1 + (size_t)s * HIDDEN);
    float* op = agg1 + (size_t)d * HIDDEN;
    #pragma unroll
    for (int q = 0; q < 4; ++q) {
        float4 v = hp[q];
        atomicAdd(op + 4 * q + 0, v.x * w);
        atomicAdd(op + 4 * q + 1, v.y * w);
        atomicAdd(op + 4 * q + 2, v.z * w);
        atomicAdd(op + 4 * q + 3, v.w * w);
    }
}

// --------------------------------- bias+ReLU, layer-2 linear, self-loop -----
__global__ __launch_bounds__(256) void k_layer2(const float* __restrict__ agg1,
                                                const float* __restrict__ b1,
                                                const float* __restrict__ W2,
                                                const float* __restrict__ dinv,
                                                float* __restrict__ h2,
                                                float* __restrict__ agg2) {
    int n = blockIdx.x * 256 + threadIdx.x;
    if (n >= N_NODES) return;
    float z[HIDDEN];
    #pragma unroll
    for (int f = 0; f < HIDDEN; ++f)
        z[f] = fmaxf(agg1[(size_t)n * HIDDEN + f] + b1[f], 0.0f);
    const float dv = dinv[n];
    const float d2 = dv * dv;
    #pragma unroll
    for (int c = 0; c < N_CLASSES; ++c) {
        float acc = 0.0f;
        #pragma unroll
        for (int f = 0; f < HIDDEN; ++f)
            acc += z[f] * W2[f * N_CLASSES + c];
        h2[(size_t)n * 8 + c]   = acc;           // stride-8 padded
        agg2[(size_t)n * 8 + c] = acc * d2;      // self-loop contribution
    }
}

// --------------------------------------------------- layer-2 edge scatter ---
__global__ __launch_bounds__(256) void k_scatter2(const int* __restrict__ src,
                                                  const int* __restrict__ dst,
                                                  const float* __restrict__ dinv,
                                                  const float* __restrict__ h2,
                                                  float* __restrict__ agg2) {
    int e = blockIdx.x * 256 + threadIdx.x;
    if (e >= N_EDGES) return;
    const int s = src[e], d = dst[e];
    const float w = dinv[s] * dinv[d];
    const float* hp = h2 + (size_t)s * 8;
    float* op = agg2 + (size_t)d * 8;
    #pragma unroll
    for (int c = 0; c < N_CLASSES; ++c)
        atomicAdd(op + c, hp[c] * w);
}

// --------------------------------------------- bias + log_softmax output ----
__global__ __launch_bounds__(256) void k_finalize(const float* __restrict__ agg2,
                                                  const float* __restrict__ b2,
                                                  float* __restrict__ out) {
    int n = blockIdx.x * 256 + threadIdx.x;
    if (n >= N_NODES) return;
    float v[N_CLASSES];
    float m = -1e30f;
    #pragma unroll
    for (int c = 0; c < N_CLASSES; ++c) {
        v[c] = agg2[(size_t)n * 8 + c] + b2[c];
        m = fmaxf(m, v[c]);
    }
    float s = 0.0f;
    #pragma unroll
    for (int c = 0; c < N_CLASSES; ++c)
        s += __expf(v[c] - m);
    const float l = __logf(s);
    #pragma unroll
    for (int c = 0; c < N_CLASSES; ++c)
        out[(size_t)n * N_CLASSES + c] = v[c] - m - l;
}

// ---------------------------------------------------------------------------
extern "C" void kernel_launch(void* const* d_in, const int* in_sizes, int n_in,
                              void* d_out, int out_size, void* d_ws, size_t ws_size,
                              hipStream_t stream) {
    const float* x  = (const float*)d_in[0];
    const int*   ei = (const int*)d_in[1];            // [2, E] flat
    const float* W1 = (const float*)d_in[2];
    const float* b1 = (const float*)d_in[3];
    const float* W2 = (const float*)d_in[4];
    const float* b2 = (const float*)d_in[5];
    const int* src = ei;
    const int* dst = ei + N_EDGES;
    float* out = (float*)d_out;

    // workspace layout (floats)
    float* ws   = (float*)d_ws;
    float* dinv = ws;                                 // N_NODES (deg -> dinv)
    float* h1   = dinv + 100352;                      // N_NODES*16
    float* agg1 = h1 + (size_t)N_NODES * HIDDEN;      // N_NODES*16
    float* h2   = agg1 + (size_t)N_NODES * HIDDEN;    // N_NODES*8 (padded)
    float* agg2 = h2 + (size_t)N_NODES * 8;           // N_NODES*8 (padded)

    const int nodeBlocks  = (N_NODES + 255) / 256;
    const int edgeBlocks  = (N_EDGES + 255) / 256;
    const int featBlocks  = (N_NODES * HIDDEN + 255) / 256;
    const int tileBlocks  = (N_NODES / 16 + 7) / 8;   // 8 waves (tiles) / block

    k_init_deg  <<<nodeBlocks, 256, 0, stream>>>(dinv);
    k_deg_accum <<<edgeBlocks, 256, 0, stream>>>(dst, dinv);
    k_deg_rsqrt <<<nodeBlocks, 256, 0, stream>>>(dinv);
    k_gemm1_wmma<<<tileBlocks, 256, 0, stream>>>(x, W1, h1);
    k_init_agg1 <<<featBlocks, 256, 0, stream>>>(h1, dinv, agg1);
    k_scatter1  <<<edgeBlocks, 256, 0, stream>>>(src, dst, dinv, h1, agg1);
    k_layer2    <<<nodeBlocks, 256, 0, stream>>>(agg1, b1, W2, dinv, h2, agg2);
    k_scatter2  <<<edgeBlocks, 256, 0, stream>>>(src, dst, dinv, h2, agg2);
    k_finalize  <<<nodeBlocks, 256, 0, stream>>>(agg2, b2, out);
}
